// SeqModel_34342558499102
// MI455X (gfx1250) — matrix operand order repaired and checked by reading, not compiled, not run on hardware
//
#include <hip/hip_runtime.h>
#include <hip/hip_bf16.h>
#include <stdint.h>

#define S_LEN   4096
#define HID     1024
#define G4      4096      // 4*H
#define IN_DIM  7
#define FC1_DIM 56
#define LBL_DIM 7
#define K1      2048      // 2*H
#define BN_EPS  1e-5f

typedef __attribute__((ext_vector_type(16))) __bf16 v16bf;
typedef __attribute__((ext_vector_type(8)))  __bf16 v8bf;
typedef __attribute__((ext_vector_type(8)))  float  v8f;

__device__ inline unsigned short f2bf(float f) {
  unsigned u = __float_as_uint(f);
  u += 0x7FFFu + ((u >> 16) & 1u);          // round-to-nearest-even
  return (unsigned short)(u >> 16);
}
__device__ inline float bf2f(unsigned short u) {
  return __uint_as_float(((unsigned)u) << 16);
}
__device__ inline float sigm(float x) { return 1.0f / (1.0f + __expf(-x)); }

// CDNA5 async global->LDS copy (no VGPR staging; tracked by ASYNCcnt).
// lds_addr: wave-relative LDS byte address (low 32 bits of flat shared ptr).
__device__ inline void async_copy_b128(unsigned lds_addr, const void* gaddr) {
  asm volatile("global_load_async_to_lds_b128 %0, %1, off"
               :: "v"(lds_addr), "v"(gaddr)
               : "memory");
}
__device__ inline void wait_async0() {
  asm volatile("s_wait_asynccnt 0x0" ::: "memory");
}
__device__ inline unsigned lds_addr_of(const void* p) {
  return (unsigned)(uintptr_t)p;   // flat shared ptr: addr[31:0] == LDS offset
}

// ---------------------------------------------------------------- utilities
__global__ void zero_u32(unsigned* p, int n) {
  int i = threadIdx.x;
  if (i < n) p[i] = 0u;
}

__global__ void f32_to_bf16(const float* __restrict__ src,
                            unsigned short* __restrict__ dst, size_t n) {
  size_t i = (size_t)blockIdx.x * blockDim.x + threadIdx.x;
  size_t stride = (size_t)gridDim.x * blockDim.x;
  for (; i < n; i += stride) dst[i] = f2bf(src[i]);
}

// xW = x[S,7] @ W^T[7,4096] + b  (tiny K, write-bandwidth bound)
__global__ void input_proj7(const float* __restrict__ x,
                            const float* __restrict__ W,
                            const float* __restrict__ b,
                            float* __restrict__ out) {
  size_t idx = (size_t)blockIdx.x * blockDim.x + threadIdx.x;
  if (idx >= (size_t)S_LEN * G4) return;
  int t = (int)(idx >> 12);
  int r = (int)(idx & 4095);
  const float* xr = x + (size_t)t * IN_DIM;
  const float* wr = W + (size_t)r * IN_DIM;
  float acc = b[r];
#pragma unroll
  for (int k = 0; k < IN_DIM; ++k) acc = fmaf(xr[k], wr[k], acc);
  out[idx] = acc;
}

// ------------------------------------------------------- WMMA bf16 GEMM
// C[M,N] = A[M,K](bf16) * B[N,K]^T(bf16) + bias[N]   (fp32 accumulate)
// Block tile 128x128, BK=32, 8 waves (4x2), each wave 32x64 = 2x4 WMMA frags.
// K-tiles double-buffered in LDS via async global->LDS copies (ASYNCcnt).
__global__ __launch_bounds__(256) void wmma_gemm_bias(
    const unsigned short* __restrict__ A,
    const unsigned short* __restrict__ B,
    const float* __restrict__ bias,
    float* __restrict__ C,
    int M, int N, int K) {
  (void)M;
  __shared__ unsigned short lA[2][128 * 32];
  __shared__ unsigned short lB[2][128 * 32];
  const int tid   = threadIdx.x;
  const int lane  = tid & 31;
  const int wave  = tid >> 5;
  const int waveM = wave >> 1;   // 0..3
  const int waveN = wave & 1;    // 0..1
  const int tileM = blockIdx.y * 128;
  const int tileN = blockIdx.x * 128;

  // per-thread 16B chunk coordinates (2 chunks each for A and B tiles)
  const int ch0 = tid,        r0 = ch0 >> 2, c0 = (ch0 & 3) * 8;
  const int ch1 = tid + 256,  r1 = ch1 >> 2, c1 = (ch1 & 3) * 8;

  v8f acc[2][4];
#pragma unroll
  for (int j = 0; j < 4; ++j) {
    const int col = tileN + waveN * 64 + j * 16 + (lane & 15);
    const float bv = bias[col];
#pragma unroll
    for (int i = 0; i < 2; ++i)
#pragma unroll
      for (int r = 0; r < 8; ++r) acc[i][j][r] = bv;
  }

  const int KT = K / 32;

  // prologue: stage K-tile 0
  async_copy_b128(lds_addr_of(&lA[0][ch0 * 8]), A + (size_t)(tileM + r0) * K + c0);
  async_copy_b128(lds_addr_of(&lA[0][ch1 * 8]), A + (size_t)(tileM + r1) * K + c1);
  async_copy_b128(lds_addr_of(&lB[0][ch0 * 8]), B + (size_t)(tileN + r0) * K + c0);
  async_copy_b128(lds_addr_of(&lB[0][ch1 * 8]), B + (size_t)(tileN + r1) * K + c1);
  wait_async0();
  __syncthreads();

  for (int kt = 0; kt < KT; ++kt) {
    const int cur = kt & 1, nxt = cur ^ 1;
    if (kt + 1 < KT) {   // prefetch next K-tile straight into LDS (no VGPRs)
      const int kk = (kt + 1) * 32;
      async_copy_b128(lds_addr_of(&lA[nxt][ch0 * 8]), A + (size_t)(tileM + r0) * K + kk + c0);
      async_copy_b128(lds_addr_of(&lA[nxt][ch1 * 8]), A + (size_t)(tileM + r1) * K + kk + c1);
      async_copy_b128(lds_addr_of(&lB[nxt][ch0 * 8]), B + (size_t)(tileN + r0) * K + kk + c0);
      async_copy_b128(lds_addr_of(&lB[nxt][ch1 * 8]), B + (size_t)(tileN + r1) * K + kk + c1);
    }
    // A fragments: 16x32, lane&15 = M row; K chunks {0..7,16..23}/{8..15,24..31}
    v16bf af[2];
#pragma unroll
    for (int i = 0; i < 2; ++i) {
      const int row = waveM * 32 + i * 16 + (lane & 15);
      const int off = (lane >> 4) * 8;
      const unsigned short* p = &lA[cur][row * 32];
      v8bf lo = *(const v8bf*)(p + off);
      v8bf hi = *(const v8bf*)(p + off + 16);
#pragma unroll
      for (int e = 0; e < 8; ++e) { af[i][e] = lo[e]; af[i][e + 8] = hi[e]; }
    }
    // B fragments: 32x16, lane&15 = N col; K chunks {0..7,16..23}/{8..15,24..31}
    v16bf bfrg[4];
#pragma unroll
    for (int j = 0; j < 4; ++j) {
      const int n  = waveN * 64 + j * 16 + (lane & 15);
      const int k0 = (lane >> 4) * 8;
      const unsigned short* p = &lB[cur][n * 32 + k0];
      v8bf lo = *(const v8bf*)(p);
      v8bf hi = *(const v8bf*)(p + 16);
#pragma unroll
      for (int e = 0; e < 8; ++e) { bfrg[j][e] = lo[e]; bfrg[j][e + 8] = hi[e]; }
    }
#pragma unroll
    for (int i = 0; i < 2; ++i)
#pragma unroll
      for (int j = 0; j < 4; ++j)
        acc[i][j] = __builtin_amdgcn_wmma_f32_16x16x32_bf16(
            false, af[i], false, bfrg[j], (short)0, acc[i][j], false, false);

    if (kt + 1 < KT) wait_async0();   // own async writes done before barrier
    __syncthreads();
  }

  // C/D layout: VGPR r -> M=r (lanes 0-15) / M=r+8 (lanes 16-31); N = lane&15
#pragma unroll
  for (int i = 0; i < 2; ++i)
#pragma unroll
    for (int j = 0; j < 4; ++j) {
      const int col   = tileN + waveN * 64 + j * 16 + (lane & 15);
      const int rbase = tileM + waveM * 32 + i * 16 + ((lane >> 4) * 8);
#pragma unroll
      for (int r = 0; r < 8; ++r)
        C[(size_t)(rbase + r) * N + col] = acc[i][j][r];
    }
}

// ------------------------------------------- persistent cooperative LSTM scan
#define SCAN_NWG   128
#define SCAN_UNITS 8     // hidden units per WG; 4 gate rows each -> 64KB LDS

struct ScanDir {
  const float* xw;            // [S, 4H] precomputed input projection (+bias)
  const unsigned short* whh;  // [4H, H] bf16
  float* h;                   // [2, H] ping-pong hidden state (global/L2)
  float* out32;               // f32 output column base (ld = 2H) or null
  unsigned short* out16;      // bf16 output column base (ld = 2H) or null
  unsigned* bar;              // monotonic grid-barrier counter
  int reverse;
};

__device__ inline void sync_grid(unsigned* bar, unsigned target, int tid) {
  __threadfence();
  __syncthreads();
  if (tid == 0) {
    atomicAdd(bar, 1u);
    while (atomicAdd(bar, 0u) < target) __builtin_amdgcn_s_sleep(2);
  }
  __syncthreads();
}

__global__ __launch_bounds__(256) void lstm_scan(ScanDir d0, ScanDir d1) {
  extern __shared__ unsigned short wlds[];   // 32 rows x 1024 bf16 = 64KB
  const int tid  = threadIdx.x;
  const int lane = tid & 31;
  const int wave = tid >> 5;
  ScanDir d = (blockIdx.x < SCAN_NWG) ? d0 : d1;
  const int wg    = blockIdx.x % SCAN_NWG;
  const int ubase = wg * SCAN_UNITS;
  const int u     = ubase + wave;            // this wave's hidden unit

  // stage this WG's W_hh slice (rows {g*H + u} for g=0..3, u in slice) to LDS
  for (int it = 0; it < 16; ++it) {
    int idx = it * 2 + (tid >> 7);           // 0..31 = w*4+g
    int w = idx >> 2, g = idx & 3;
    const uint4* src = (const uint4*)(d.whh + (size_t)(g * HID + ubase + w) * HID);
    uint4* dst = (uint4*)(wlds + idx * HID);
    dst[tid & 127] = src[tid & 127];
  }
  if (tid < SCAN_UNITS) d.h[ubase + tid] = 0.0f;   // zero h ping buffer
  __syncthreads();
  sync_grid(d.bar, SCAN_NWG, tid);

  float creg = 0.0f;   // cell state (valid on lane 0)

  for (int s = 0; s < S_LEN; ++s) {
    const int t = d.reverse ? (S_LEN - 1 - s) : s;
    const float* hsrc = d.h + (s & 1) * HID;
    float* hdst       = d.h + ((s + 1) & 1) * HID;

    float hreg[32];
    {
      const float4* hp = (const float4*)(hsrc + lane * 32);
#pragma unroll
      for (int j2 = 0; j2 < 8; ++j2) {
        float4 v = hp[j2];
        hreg[4 * j2 + 0] = v.x; hreg[4 * j2 + 1] = v.y;
        hreg[4 * j2 + 2] = v.z; hreg[4 * j2 + 3] = v.w;
      }
    }
    float gdot[4];
#pragma unroll
    for (int g = 0; g < 4; ++g) {
      const unsigned short* wr = wlds + (wave * 4 + g) * HID + lane * 32;
      float acc = 0.0f;
#pragma unroll
      for (int k = 0; k < 32; ++k) acc = fmaf(bf2f(wr[k]), hreg[k], acc);
#pragma unroll
      for (int off = 16; off > 0; off >>= 1) acc += __shfl_down(acc, off, 32);
      gdot[g] = acc;   // valid on lane 0
    }
    if (lane == 0) {
      const float* xwrow = d.xw + (size_t)t * G4;
      float zi = xwrow[u            ] + gdot[0];
      float zf = xwrow[HID     + u  ] + gdot[1];
      float zg = xwrow[2 * HID + u  ] + gdot[2];
      float zo = xwrow[3 * HID + u  ] + gdot[3];
      creg = sigm(zf) * creg + sigm(zi) * tanhf(zg);
      float hn = sigm(zo) * tanhf(creg);
      hdst[u] = hn;
      if (d.out32) d.out32[(size_t)t * K1 + u] = hn;
      if (d.out16) d.out16[(size_t)t * K1 + u] = f2bf(hn);
    }
    sync_grid(d.bar, (unsigned)SCAN_NWG * (unsigned)(s + 2), tid);
  }
}

// --------------------------------------------------------------- head kernels
__global__ __launch_bounds__(256) void fc1_kernel(
    const float* __restrict__ X,   // [S, 2H]
    const float* __restrict__ W,   // [56, 2H]
    const float* __restrict__ b,
    float* __restrict__ out) {     // [S, 56]
  __shared__ float sx[K1];
  const int s = blockIdx.x;
  const int tid = threadIdx.x, lane = tid & 31, wave = tid >> 5;
  for (int k = tid; k < K1; k += 256) sx[k] = X[(size_t)s * K1 + k];
  __syncthreads();
  for (int j = wave; j < FC1_DIM; j += 8) {
    const float* wr = W + (size_t)j * K1;
    float acc = 0.0f;
    for (int k = lane; k < K1; k += 32) acc = fmaf(sx[k], wr[k], acc);
#pragma unroll
    for (int off = 16; off > 0; off >>= 1) acc += __shfl_down(acc, off, 32);
    if (lane == 0) out[(size_t)s * FC1_DIM + j] = acc + b[j];
  }
}

__global__ __launch_bounds__(256) void bn_stats(const float* __restrict__ h1,
                                                float* __restrict__ meanv,
                                                float* __restrict__ varv) {
  __shared__ float ssum[256], ssq[256];
  const int j = blockIdx.x;
  const int tid = threadIdx.x;
  float s = 0.0f, q = 0.0f;
  for (int r = tid; r < S_LEN; r += 256) {
    float v = h1[(size_t)r * FC1_DIM + j];
    s += v; q = fmaf(v, v, q);
  }
  ssum[tid] = s; ssq[tid] = q; __syncthreads();
  for (int off = 128; off > 0; off >>= 1) {
    if (tid < off) { ssum[tid] += ssum[tid + off]; ssq[tid] += ssq[tid + off]; }
    __syncthreads();
  }
  if (tid == 0) {
    float m = ssum[0] / (float)S_LEN;
    meanv[j] = m;
    varv[j] = ssq[0] / (float)S_LEN - m * m;   // biased, training-mode BN
  }
}

__global__ __launch_bounds__(256) void head_kernel(
    const float* __restrict__ h1, const float* __restrict__ meanv,
    const float* __restrict__ varv, const float* __restrict__ gamma,
    const float* __restrict__ beta, const float* __restrict__ W2,
    const float* __restrict__ b2, const float* __restrict__ inp,
    float* __restrict__ out) {
  int s = blockIdx.x * blockDim.x + threadIdx.x;
  if (s >= S_LEN) return;
  float a[FC1_DIM];
#pragma unroll 8
  for (int j = 0; j < FC1_DIM; ++j) {
    float v = h1[(size_t)s * FC1_DIM + j];
    float bn = (v - meanv[j]) * rsqrtf(varv[j] + BN_EPS) * gamma[j] + beta[j];
    a[j] = fmaxf(bn, 0.0f);
  }
#pragma unroll
  for (int l = 0; l < LBL_DIM; ++l) {
    float acc = b2[l];
    for (int j = 0; j < FC1_DIM; ++j) acc = fmaf(a[j], W2[l * FC1_DIM + j], acc);
    out[(size_t)s * LBL_DIM + l] = acc + inp[(size_t)s * LBL_DIM + l];
  }
}

// -------------------------------------------------------------------- launch
extern "C" void kernel_launch(void* const* d_in, const int* in_sizes, int n_in,
                              void* d_out, int out_size, void* d_ws, size_t ws_size,
                              hipStream_t stream) {
  (void)in_sizes; (void)n_in; (void)out_size; (void)ws_size;
  const float* inputs  = (const float*)d_in[0];
  const float* Wih_l0f = (const float*)d_in[1];
  const float* Whh_l0f = (const float*)d_in[2];
  const float* b_l0f   = (const float*)d_in[3];
  const float* Wih_l0b = (const float*)d_in[4];
  const float* Whh_l0b = (const float*)d_in[5];
  const float* b_l0b   = (const float*)d_in[6];
  const float* Wih_l1f = (const float*)d_in[7];
  const float* Whh_l1f = (const float*)d_in[8];
  const float* b_l1f   = (const float*)d_in[9];
  const float* Wih_l1b = (const float*)d_in[10];
  const float* Whh_l1b = (const float*)d_in[11];
  const float* b_l1b   = (const float*)d_in[12];
  // d_in[13..15] = W_attn, b_attn, v_attn: softmax over batch==1 axis is
  // identically 1 -> context == lstm_out; attention block is eliminated.
  const float* W_fc1 = (const float*)d_in[16];
  const float* b_fc1 = (const float*)d_in[17];
  const float* gamma = (const float*)d_in[18];
  const float* beta  = (const float*)d_in[19];
  const float* W_fc2 = (const float*)d_in[20];
  const float* b_fc2 = (const float*)d_in[21];
  float* out = (float*)d_out;

  char* w = (char*)d_ws;
  auto alloc = [&](size_t bytes) -> char* {
    char* p = w; w += (bytes + 255) & ~(size_t)255; return p;
  };
  float* xwA            = (float*)alloc((size_t)S_LEN * G4 * 4);   // dirF xW (both layers)
  float* xwB            = (float*)alloc((size_t)S_LEN * G4 * 4);   // dirB xW (both layers)
  unsigned short* x1    = (unsigned short*)alloc((size_t)S_LEN * K1 * 2);  // layer0 out, bf16
  float* lstm_out       = (float*)alloc((size_t)S_LEN * K1 * 4);
  unsigned short* whhbf[4];
  for (int i = 0; i < 4; ++i)
    whhbf[i] = (unsigned short*)alloc((size_t)G4 * HID * 2);
  unsigned short* wih1f = (unsigned short*)alloc((size_t)G4 * K1 * 2);
  unsigned short* wih1b = (unsigned short*)alloc((size_t)G4 * K1 * 2);
  float* hbuf           = (float*)alloc((size_t)4 * HID * 4);      // 2 dirs x ping/pong
  float* h1fc           = (float*)alloc((size_t)S_LEN * FC1_DIM * 4);
  float* meanv          = (float*)alloc(FC1_DIM * 4);
  float* varv           = (float*)alloc(FC1_DIM * 4);
  unsigned* bars        = (unsigned*)alloc(16 * 4);

  zero_u32<<<1, 32, 0, stream>>>(bars, 16);

  const float* whsrc[4] = {Whh_l0f, Whh_l0b, Whh_l1f, Whh_l1b};
  for (int i = 0; i < 4; ++i)
    f32_to_bf16<<<512, 256, 0, stream>>>(whsrc[i], whhbf[i], (size_t)G4 * HID);
  f32_to_bf16<<<1024, 256, 0, stream>>>(Wih_l1f, wih1f, (size_t)G4 * K1);
  f32_to_bf16<<<1024, 256, 0, stream>>>(Wih_l1b, wih1b, (size_t)G4 * K1);

  const int np = (S_LEN * G4 + 255) / 256;
  input_proj7<<<np, 256, 0, stream>>>(inputs, Wih_l0f, b_l0f, xwA);
  input_proj7<<<np, 256, 0, stream>>>(inputs, Wih_l0b, b_l0b, xwB);

  // layer-0 scan: both directions in one cooperative launch, W_hh LDS-resident
  ScanDir d0{xwA, whhbf[0], hbuf,           nullptr, x1,       bars + 0, 0};
  ScanDir d1{xwB, whhbf[1], hbuf + 2 * HID, nullptr, x1 + HID, bars + 1, 1};
  lstm_scan<<<2 * SCAN_NWG, 256, 64 * 1024, stream>>>(d0, d1);

  // layer-1 input projections: big WMMA bf16 GEMMs (re-use xW buffers)
  dim3 gg(G4 / 128, S_LEN / 128);
  wmma_gemm_bias<<<gg, 256, 0, stream>>>(x1, wih1f, b_l1f, xwA, S_LEN, G4, K1);
  wmma_gemm_bias<<<gg, 256, 0, stream>>>(x1, wih1b, b_l1b, xwB, S_LEN, G4, K1);

  ScanDir e0{xwA, whhbf[2], hbuf,           lstm_out,       nullptr, bars + 2, 0};
  ScanDir e1{xwB, whhbf[3], hbuf + 2 * HID, lstm_out + HID, nullptr, bars + 3, 1};
  lstm_scan<<<2 * SCAN_NWG, 256, 64 * 1024, stream>>>(e0, e1);

  fc1_kernel<<<S_LEN, 256, 0, stream>>>(lstm_out, W_fc1, b_fc1, h1fc);
  bn_stats<<<FC1_DIM, 256, 0, stream>>>(h1fc, meanv, varv);
  head_kernel<<<(S_LEN + 255) / 256, 256, 0, stream>>>(
      h1fc, meanv, varv, gamma, beta, W_fc2, b_fc2, inputs, out);
}
